// GCM_85272280695503
// MI455X (gfx1250) — matrix operand order repaired and checked by reading, not compiled, not run on hardware
//
#include <hip/hip_runtime.h>
#include <hip/hip_bf16.h>
#include <math.h>

// ---------------------------------------------------------------------------
// Shapes (fixed by the reference)
// ---------------------------------------------------------------------------
#define BB 4
#define CC 64
#define HH 128
#define WW 128
#define HP 64
#define WP 64
#define NTOK (HP * WP)          // 4096 tokens per batch
#define MTOT (BB * NTOK)        // 16384 rows total
#define SCALE_ATTN 0.125f       // 64^-0.5

typedef __attribute__((ext_vector_type(16))) _Float16 v16h;
typedef __attribute__((ext_vector_type(8)))  float    v8f;
typedef __attribute__((ext_vector_type(4)))  int      v4i;

union F16x16 { v16h v; uint4 u[2]; _Float16 h[16]; };

// gfx1250 async global->LDS path (ASYNCcnt-tracked), probe-verified arg types.
#if defined(__AMDGCN__) && __has_builtin(__builtin_amdgcn_global_load_async_to_lds_b128)
#define ASYNC_LDS 1
typedef __attribute__((address_space(1))) v4i* gv4i_p;  // global int4*
typedef __attribute__((address_space(3))) v4i* lv4i_p;  // LDS int4*
#else
#define ASYNC_LDS 0
#endif

static __device__ __forceinline__ void wait_asynccnt0() {
#if ASYNC_LDS
#if __has_builtin(__builtin_amdgcn_s_wait_asynccnt)
  __builtin_amdgcn_s_wait_asynccnt(0);
#else
  asm volatile("s_wait_asynccnt 0" ::: "memory");
#endif
#endif
}

// Cross-lane XOR reduce step within 16-lane rows: DPP row_xmask (pure VALU)
// instead of ds_bpermute (LDS pipeline + dscnt waits).
#if defined(__AMDGCN__) && __has_builtin(__builtin_amdgcn_update_dpp)
template <int MASK>
static __device__ __forceinline__ float xor_lane_f32(float x) {
  int i = __builtin_bit_cast(int, x);
  int r = __builtin_amdgcn_update_dpp(i, i, 0x160 | MASK, 0xf, 0xf, false);
  return __builtin_bit_cast(float, r);
}
#else
template <int MASK>
static __device__ __forceinline__ float xor_lane_f32(float x) {
  return __shfl_xor(x, MASK, 32);
}
#endif

static __device__ __forceinline__ float rowmax16(float x) {
  x = fmaxf(x, xor_lane_f32<1>(x));
  x = fmaxf(x, xor_lane_f32<2>(x));
  x = fmaxf(x, xor_lane_f32<4>(x));
  x = fmaxf(x, xor_lane_f32<8>(x));
  return x;
}

static __device__ __forceinline__ v8f wmma16(v16h a, v16h b, v8f c) {
  // (neg_a, A, neg_b, B, c_mod, C, reuse_a, reuse_b)
  return __builtin_amdgcn_wmma_f32_16x16x32_f16(false, a, false, b, (short)0, c,
                                                false, false);
}

// Load 16 contiguous halfs (B-matrix fragment: lane n = lane&15, k = 16*hi + h)
static __device__ __forceinline__ v16h ldb16(const _Float16* p) {
  F16x16 f;
  f.u[0] = *(const uint4*)(p);
  f.u[1] = *(const uint4*)(p + 8);
  return f.v;
}

static __device__ __forceinline__ float gelu_exact(float x) {
  return 0.5f * x * (1.0f + erff(x * 0.70710678118654752f));
}

// ---------------------------------------------------------------------------
// 1) AvgPool2d(2,2):  x [B,C,128,128] -> xp [B,C,64,64] (channel-major f32)
// ---------------------------------------------------------------------------
__global__ __launch_bounds__(256) void avgpool_kernel(const float* __restrict__ x,
                                                      float* __restrict__ xp) {
  int idx = blockIdx.x * 256 + threadIdx.x;       // B*C*64*64
  int w  = idx & 63;
  int h  = (idx >> 6) & 63;
  int bc = idx >> 12;
  const float* p = x + ((long)bc * HH + h * 2) * WW + w * 2;
  xp[idx] = 0.25f * (p[0] + p[1] + p[WW] + p[WW + 1]);
}

// ---------------------------------------------------------------------------
// 2) Global average pool over spatial: gap [B*C]
// ---------------------------------------------------------------------------
__global__ __launch_bounds__(256) void gap_kernel(const float* __restrict__ xp,
                                                  float* __restrict__ gap) {
  __shared__ float red[256];
  int bc = blockIdx.x, tid = threadIdx.x;
  float s = 0.0f;
  for (int i = tid; i < NTOK; i += 256) s += xp[(long)bc * NTOK + i];
  red[tid] = s;
  __syncthreads();
  for (int o = 128; o > 0; o >>= 1) {
    if (tid < o) red[tid] += red[tid + o];
    __syncthreads();
  }
  if (tid == 0) gap[bc] = red[0] * (1.0f / (float)NTOK);
}

// ---------------------------------------------------------------------------
// 3) Tokenize: xg = gap*xp, xpT = xp  re-laid out as [B,N,C] f16 token-major
// ---------------------------------------------------------------------------
__global__ __launch_bounds__(256) void tokenize_kernel(const float* __restrict__ xp,
                                                       const float* __restrict__ gap,
                                                       _Float16* __restrict__ xg,
                                                       _Float16* __restrict__ xpT) {
  int idx = blockIdx.x * 256 + threadIdx.x;       // [b][c][n]
  int n = idx & (NTOK - 1);
  int bc = idx >> 12;
  int b = bc >> 6, c = bc & 63;
  float v = xp[idx];
  long o = ((long)b * NTOK + n) * CC + c;
  xg[o]  = (_Float16)(gap[bc] * v);
  xpT[o] = (_Float16)v;
}

// ---------------------------------------------------------------------------
// 4) Generic WMMA GEMM: Y[M,64] = X[M,64] @ W^T + bias (+ resid)
//    X token-major f16, W f32 [o][c] (Linear / 1x1-conv convention).
//    Compile-time config => branch-free epilogue.
// ---------------------------------------------------------------------------
template <bool RESID, bool OUTF, bool OUTH>
__global__ __launch_bounds__(256) void gemm64_kernel(const _Float16* __restrict__ X,
                                                     const float* __restrict__ W,
                                                     const float* __restrict__ bias,
                                                     const float* __restrict__ resid,
                                                     float* __restrict__ Yf,
                                                     _Float16* __restrict__ Yh) {
  __shared__ _Float16 Wlds[CC * CC];              // 8 KB, W as f16 [o][c]
  const int tid = threadIdx.x;
  for (int i = tid; i < CC * CC; i += 256) Wlds[i] = (_Float16)W[i];
  __syncthreads();

  const int wave = tid >> 5, lane = tid & 31;
  const int r = lane & 15, hi = lane >> 4;
  const long rowbase = (long)blockIdx.x * 128 + wave * 16;

  // A fragments (16x32 each, K-chunks c=0..31 and c=32..63)
  const _Float16* xrow = X + (rowbase + r) * CC;
  F16x16 a0, a1;
  a0.u[0] = *(const uint4*)(xrow + 16 * hi);
  a0.u[1] = *(const uint4*)(xrow + 8 + 16 * hi);
  a1.u[0] = *(const uint4*)(xrow + 32 + 16 * hi);
  a1.u[1] = *(const uint4*)(xrow + 40 + 16 * hi);

#pragma unroll
  for (int ot = 0; ot < 4; ot++) {
    const int col = 16 * ot + r;                  // output channel o
    v16h b0 = ldb16(&Wlds[col * CC + 16 * hi]);        // c = 16*hi + h
    v16h b1 = ldb16(&Wlds[col * CC + 32 + 16 * hi]);   // c = 32 + 16*hi + h
    v8f acc = {};
    acc = wmma16(a0.v, b0, acc);
    acc = wmma16(a1.v, b1, acc);
    const float bv = bias[col];
    const long base = (rowbase + 8 * hi) * CC + col;
#pragma unroll
    for (int e = 0; e < 8; e++) {
      float y = acc[e] + bv;
      if (RESID) y += resid[base + (long)e * CC];
      if (OUTF) Yf[base + (long)e * CC] = y;
      if (OUTH) Yh[base + (long)e * CC] = (_Float16)y;
    }
  }
}

// ---------------------------------------------------------------------------
// 5) Flash attention: O = softmax(Q K^T * scale) V,  per batch N=4096, C=64
//    Wave = 16 query rows; block = 8 waves = 128 queries; 32-key inner tiles.
//    - K tile staged via async global->LDS (ASYNCcnt)
//    - row max via DPP row_xmask (VALU), row sum via ones-matrix WMMA
// ---------------------------------------------------------------------------
__global__ __launch_bounds__(256) void attn_kernel(const _Float16* __restrict__ Q,
                                                   const _Float16* __restrict__ K,
                                                   const _Float16* __restrict__ V,
                                                   _Float16* __restrict__ Oh,
                                                   float scale) {
  __shared__ _Float16 Klds[32 * CC];              // K tile  [key][c]   4 KB
  __shared__ _Float16 VTlds[CC * 32];             // V^T tile [c][key]  4 KB
  __shared__ _Float16 Plds[8][16 * 32];           // per-wave P tile    8 KB

  const int tid = threadIdx.x;
  const int wave = tid >> 5, lane = tid & 31;
  const int r = lane & 15, hi = lane >> 4;
  const long boff = (long)blockIdx.y * NTOK * CC;
  const int qbase = blockIdx.x * 128 + wave * 16;

  // Per-thread staging coordinates (8 halfs = 16 B per thread, 32x64 tile)
  const int sidx = tid * 8;
  const int srow = sidx >> 6, scol = sidx & 63;

  // All-ones B fragment: one WMMA against it yields row-sums of P in every
  // column of the C/D tile (element e = row e + 8*hi).
  F16x16 ones;
#pragma unroll
  for (int i = 0; i < 16; i++) ones.h[i] = (_Float16)1.0f;

  // Q A-fragments, kept in registers for the whole kernel
  const _Float16* qrow = Q + boff + (long)(qbase + r) * CC;
  F16x16 qa0, qa1;
  qa0.u[0] = *(const uint4*)(qrow + 16 * hi);
  qa0.u[1] = *(const uint4*)(qrow + 8 + 16 * hi);
  qa1.u[0] = *(const uint4*)(qrow + 32 + 16 * hi);
  qa1.u[1] = *(const uint4*)(qrow + 40 + 16 * hi);

  v8f o0 = {}, o1 = {}, o2 = {}, o3 = {};
  float m[8], l[8], al[8];
#pragma unroll
  for (int e = 0; e < 8; e++) { m[e] = -1.0e30f; l[e] = 0.0f; }

#pragma unroll 1
  for (int kb = 0; kb < NTOK; kb += 32) {
    __syncthreads();                              // prior-iter LDS reads done
    {                                             // cooperative K / V^T staging
      const _Float16* gk = K + boff + (long)(kb + srow) * CC + scol;
#if ASYNC_LDS
      __builtin_amdgcn_global_load_async_to_lds_b128(
          (gv4i_p)gk, (lv4i_p)&Klds[sidx], 0, 0);
#else
      *(uint4*)&Klds[sidx] = *(const uint4*)gk;
#endif
      uint4 vv = *(const uint4*)(V + boff + (long)(kb + srow) * CC + scol);
      const _Float16* vh = (const _Float16*)&vv;
#pragma unroll
      for (int i = 0; i < 8; i++) VTlds[(scol + i) * 32 + srow] = vh[i];
    }
    // Prefetch next tile while this one is consumed (global_prefetch_b8)
    if (kb + 32 < NTOK) {
      __builtin_prefetch(K + boff + (long)(kb + 32 + srow) * CC + scol, 0, 1);
      __builtin_prefetch(V + boff + (long)(kb + 32 + srow) * CC + scol, 0, 1);
    }
    wait_asynccnt0();
    __syncthreads();

    // S = Q K^T  (two 16x16 output tiles over 32 keys)
    v8f s0 = {}, s1 = {};
    {
      v16h b;
      b = ldb16(&Klds[(0 + r)  * CC + 0  + 16 * hi]); s0 = wmma16(qa0.v, b, s0);
      b = ldb16(&Klds[(0 + r)  * CC + 32 + 16 * hi]); s0 = wmma16(qa1.v, b, s0);
      b = ldb16(&Klds[(16 + r) * CC + 0  + 16 * hi]); s1 = wmma16(qa0.v, b, s1);
      b = ldb16(&Klds[(16 + r) * CC + 32 + 16 * hi]); s1 = wmma16(qa1.v, b, s1);
    }

    // Online softmax max + exp (row = e + 8*hi); sums done by WMMA below
#pragma unroll
    for (int e = 0; e < 8; e++) {
      float x0 = s0[e] * scale, x1 = s1[e] * scale;
      float mx = rowmax16(fmaxf(x0, x1));
      float mn = fmaxf(m[e], mx);
      al[e] = __expf(m[e] - mn);
      m[e] = mn;
      int row = e + 8 * hi;
      Plds[wave][row * 32 + r]      = (_Float16)__expf(x0 - mn);
      Plds[wave][row * 32 + 16 + r] = (_Float16)__expf(x1 - mn);
    }
    __syncthreads();                              // C/D->A layout round trip

    // Reload P as A fragment
    F16x16 pa;
    const _Float16* prow = &Plds[wave][r * 32];
    pa.u[0] = *(const uint4*)(prow + 16 * hi);
    pa.u[1] = *(const uint4*)(prow + 8 + 16 * hi);

    // Row-sums of P via ones-matrix WMMA (replaces 32 ds_bpermute reductions)
    v8f rs = {};
    rs = wmma16(pa.v, ones.v, rs);

    // Rescale running state, then accumulate O += P V
#pragma unroll
    for (int e = 0; e < 8; e++) {
      l[e] = l[e] * al[e] + rs[e];
      o0[e] *= al[e]; o1[e] *= al[e]; o2[e] *= al[e]; o3[e] *= al[e];
    }
    {
      v16h b;
      b = ldb16(&VTlds[(0  + r) * 32 + 16 * hi]); o0 = wmma16(pa.v, b, o0);
      b = ldb16(&VTlds[(16 + r) * 32 + 16 * hi]); o1 = wmma16(pa.v, b, o1);
      b = ldb16(&VTlds[(32 + r) * 32 + 16 * hi]); o2 = wmma16(pa.v, b, o2);
      b = ldb16(&VTlds[(48 + r) * 32 + 16 * hi]); o3 = wmma16(pa.v, b, o3);
    }
  }

#pragma unroll
  for (int e = 0; e < 8; e++) {
    long row = qbase + e + 8 * hi;
    float inv = 1.0f / l[e];
    Oh[boff + row * CC + 0  + r] = (_Float16)(o0[e] * inv);
    Oh[boff + row * CC + 16 + r] = (_Float16)(o1[e] * inv);
    Oh[boff + row * CC + 32 + r] = (_Float16)(o2[e] * inv);
    Oh[boff + row * CC + 48 + r] = (_Float16)(o3[e] * inv);
  }
}

// ---------------------------------------------------------------------------
// 6) Depthwise 5x1 / 1x5 strip conv + bias + GELU -> token-major f16
// ---------------------------------------------------------------------------
__global__ __launch_bounds__(256) void dwstrip_kernel(const float* __restrict__ xp,
                                                      const float* __restrict__ w5,
                                                      const float* __restrict__ bias,
                                                      int vertical,
                                                      _Float16* __restrict__ outT) {
  int idx = blockIdx.x * 256 + threadIdx.x;       // [b][c][h][w]
  int wx = idx & 63, hy = (idx >> 6) & 63;
  int bc = idx >> 12, c = bc & 63, b = bc >> 6;
  const float* base = xp + (long)bc * NTOK;
  float acc = bias[c];
#pragma unroll
  for (int t = 0; t < 5; t++) {
    int hh = hy + (vertical ? (t - 2) : 0);
    int ww = wx + (vertical ? 0 : (t - 2));
    if (hh >= 0 && hh < HP && ww >= 0 && ww < WP)
      acc += w5[c * 5 + t] * base[hh * WP + ww];
  }
  outT[((long)b * NTOK + hy * WP + wx) * CC + c] = (_Float16)gelu_exact(acc);
}

// ---------------------------------------------------------------------------
// 7) Depthwise 3x3 + bias + GELU -> token-major f16
// ---------------------------------------------------------------------------
__global__ __launch_bounds__(256) void dw3x3_kernel(const float* __restrict__ xsp,
                                                    const float* __restrict__ w9,
                                                    const float* __restrict__ bias,
                                                    _Float16* __restrict__ outT) {
  int idx = blockIdx.x * 256 + threadIdx.x;
  int wx = idx & 63, hy = (idx >> 6) & 63;
  int bc = idx >> 12, c = bc & 63, b = bc >> 6;
  const float* base = xsp + (long)bc * NTOK;
  float acc = bias[c];
#pragma unroll
  for (int kh = 0; kh < 3; kh++) {
#pragma unroll
    for (int kw = 0; kw < 3; kw++) {
      int hh = hy + kh - 1, ww = wx + kw - 1;
      if (hh >= 0 && hh < HP && ww >= 0 && ww < WP)
        acc += w9[c * 9 + kh * 3 + kw] * base[hh * WP + ww];
    }
  }
  outT[((long)b * NTOK + hy * WP + wx) * CC + c] = (_Float16)gelu_exact(acc);
}

// ---------------------------------------------------------------------------
// 8) LayerNorm over C per token; dual outputs (token-major + channel-major)
// ---------------------------------------------------------------------------
__global__ __launch_bounds__(256) void ln_kernel(const float* __restrict__ P,
                                                 const float* __restrict__ g,
                                                 const float* __restrict__ bta,
                                                 float* __restrict__ outT,
                                                 float* __restrict__ outSp) {
  int t = blockIdx.x * 256 + threadIdx.x;         // 16384 tokens
  const float* p = P + (long)t * CC;
  float mu = 0.0f;
  for (int c = 0; c < CC; c++) mu += p[c];
  mu *= (1.0f / CC);
  float var = 0.0f;
  for (int c = 0; c < CC; c++) { float d = p[c] - mu; var += d * d; }
  var *= (1.0f / CC);
  float inv = rsqrtf(var + 1e-5f);
  int b = t >> 12, n = t & (NTOK - 1);
  for (int c = 0; c < CC; c++) {
    float y = (p[c] - mu) * inv * g[c] + bta[c];
    outT[(long)t * CC + c] = y;
    outSp[((long)b * CC + c) * NTOK + n] = y;
  }
}

// ---------------------------------------------------------------------------
// 9) Bilinear x2 upsample, align_corners=True: token-major f32 -> d_out
// ---------------------------------------------------------------------------
__global__ __launch_bounds__(256) void up2_kernel(const float* __restrict__ inT,
                                                  float* __restrict__ out) {
  int idx = blockIdx.x * 256 + threadIdx.x;       // [b][c][i][j], 128x128
  int j = idx & 127, i = (idx >> 7) & 127;
  int c = (idx >> 14) & 63, b = idx >> 20;
  const float s = 63.0f / 127.0f;
  float sy = i * s, sx = j * s;
  int y0 = (int)sy, x0 = (int)sx;
  int y1 = min(y0 + 1, HP - 1), x1 = min(x0 + 1, WP - 1);
  float wy = sy - y0, wx = sx - x0;
  const float* base = inT + (long)b * NTOK * CC + c;
  float v00 = base[(y0 * WP + x0) * CC];
  float v01 = base[(y0 * WP + x1) * CC];
  float v10 = base[(y1 * WP + x0) * CC];
  float v11 = base[(y1 * WP + x1) * CC];
  out[idx] = (1.0f - wy) * ((1.0f - wx) * v00 + wx * v01) +
             wy * ((1.0f - wx) * v10 + wx * v11);
}

// ---------------------------------------------------------------------------
// Launch
// ---------------------------------------------------------------------------
extern "C" void kernel_launch(void* const* d_in, const int* in_sizes, int n_in,
                              void* d_out, int out_size, void* d_ws, size_t ws_size,
                              hipStream_t stream) {
  const float* x      = (const float*)d_in[0];
  const float* Wq = (const float*)d_in[1],  * bq = (const float*)d_in[2];
  const float* Wk = (const float*)d_in[3],  * bk = (const float*)d_in[4];
  const float* Wv = (const float*)d_in[5],  * bv = (const float*)d_in[6];
  const float* Wl = (const float*)d_in[7],  * bl = (const float*)d_in[8];
  const float* Wo = (const float*)d_in[9],  * bo = (const float*)d_in[10];
  const float* Wp = (const float*)d_in[11], * bp = (const float*)d_in[12];
  const float* sch_dw = (const float*)d_in[13], * sch_dwb = (const float*)d_in[14];
  const float* sch_pw = (const float*)d_in[15], * sch_pwb = (const float*)d_in[16];
  const float* scv_dw = (const float*)d_in[17], * scv_dwb = (const float*)d_in[18];
  const float* scv_pw = (const float*)d_in[19], * scv_pwb = (const float*)d_in[20];
  const float* convh_w = (const float*)d_in[21], * convh_b = (const float*)d_in[22];
  const float* dsc_dw = (const float*)d_in[23], * dsc_dwb = (const float*)d_in[24];
  const float* dsc_pw = (const float*)d_in[25], * dsc_pwb = (const float*)d_in[26];
  const float* ln_g = (const float*)d_in[27], * ln_b = (const float*)d_in[28];
  float* out = (float*)d_out;

  // ---- workspace carving (all offsets 256B aligned) ----
  char* ws = (char*)d_ws;
  size_t off = 0;
  auto carve = [&](size_t bytes) {
    void* p = ws + off;
    off += (bytes + 255) & ~(size_t)255;
    return p;
  };
  const size_t NE = (size_t)MTOT * CC;            // 1,048,576 elements
  float*    xp_f     = (float*)   carve(NE * 4);  // [B,C,64,64]
  float*    gap_f    = (float*)   carve(BB * CC * 4);
  _Float16* xg_h     = (_Float16*)carve(NE * 2);  // [B,N,C]
  _Float16* xpT_h    = (_Float16*)carve(NE * 2);
  _Float16* q_h      = (_Float16*)carve(NE * 2);
  _Float16* k_h      = (_Float16*)carve(NE * 2);
  _Float16* v_h      = (_Float16*)carve(NE * 2);
  _Float16* att1_h   = (_Float16*)carve(NE * 2);
  _Float16* pl_h     = (_Float16*)carve(NE * 2);  // prompt_l (K of attn2)
  _Float16* sch_h    = (_Float16*)carve(NE * 2);
  _Float16* scv_h    = (_Float16*)carve(NE * 2);
  float*    xh_f     = (float*)   carve(NE * 4);  // x_h (f32, residual source)
  _Float16* sum_h    = (_Float16*)carve(NE * 2);  // x_h + x_v
  _Float16* ph_h     = (_Float16*)carve(NE * 2);  // prompt_h (Q of attn2)
  float*    xV_f     = (float*)   carve(NE * 4);
  _Float16* xV_h     = (_Float16*)carve(NE * 2);
  _Float16* att2_h   = (_Float16*)carve(NE * 2);
  float*    prm_f    = (float*)   carve(NE * 4);  // prompt (pre-LN)
  float*    pnT_f    = (float*)   carve(NE * 4);  // p_norm token-major
  float*    pnS_f    = (float*)   carve(NE * 4);  // p_norm channel-major
  _Float16* dsc_h    = (_Float16*)carve(NE * 2);
  float*    dout_f   = (float*)   carve(NE * 4);  // dsc output token-major
  (void)ws_size;

  const int ELEM_BLKS = (int)(NE / 256);          // 4096
  const dim3 GEMM_GRID(MTOT / 128);               // 128 blocks
  const dim3 ATTN_GRID(NTOK / 128, BB);           // (32, 4)

  // 1-3: pool, GAP, tokenize
  avgpool_kernel<<<ELEM_BLKS, 256, 0, stream>>>(x, xp_f);
  gap_kernel<<<BB * CC, 256, 0, stream>>>(xp_f, gap_f);
  tokenize_kernel<<<ELEM_BLKS, 256, 0, stream>>>(xp_f, gap_f, xg_h, xpT_h);

  // forward_lpg: q/k/v -> attn -> Wl
  gemm64_kernel<false, false, true><<<GEMM_GRID, 256, 0, stream>>>(
      xg_h, Wq, bq, nullptr, nullptr, q_h);
  gemm64_kernel<false, false, true><<<GEMM_GRID, 256, 0, stream>>>(
      xg_h, Wk, bk, nullptr, nullptr, k_h);
  gemm64_kernel<false, false, true><<<GEMM_GRID, 256, 0, stream>>>(
      xg_h, Wv, bv, nullptr, nullptr, v_h);
  attn_kernel<<<ATTN_GRID, 256, 0, stream>>>(q_h, k_h, v_h, att1_h, SCALE_ATTN);
  gemm64_kernel<false, false, true><<<GEMM_GRID, 256, 0, stream>>>(
      att1_h, Wl, bl, nullptr, nullptr, pl_h);

  // forward_hpg: strip convs -> pointwise -> sum -> convh
  dwstrip_kernel<<<ELEM_BLKS, 256, 0, stream>>>(xp_f, sch_dw, sch_dwb, 1, sch_h);
  dwstrip_kernel<<<ELEM_BLKS, 256, 0, stream>>>(xp_f, scv_dw, scv_dwb, 0, scv_h);
  gemm64_kernel<false, true, false><<<GEMM_GRID, 256, 0, stream>>>(
      sch_h, sch_pw, sch_pwb, nullptr, xh_f, nullptr);
  gemm64_kernel<true, false, true><<<GEMM_GRID, 256, 0, stream>>>(
      scv_h, scv_pw, scv_pwb, xh_f, nullptr, sum_h);
  gemm64_kernel<false, false, true><<<GEMM_GRID, 256, 0, stream>>>(
      sum_h, convh_w, convh_b, nullptr, nullptr, ph_h);

  // forward_pgc: x_V, cross-attn, Wp + residual
  gemm64_kernel<false, true, true><<<GEMM_GRID, 256, 0, stream>>>(
      xpT_h, Wo, bo, nullptr, xV_f, xV_h);
  attn_kernel<<<ATTN_GRID, 256, 0, stream>>>(ph_h, pl_h, xV_h, att2_h, SCALE_ATTN);
  gemm64_kernel<true, true, false><<<GEMM_GRID, 256, 0, stream>>>(
      att2_h, Wp, bp, xV_f, prm_f, nullptr);

  // LayerNorm, dsc (dw3x3 -> GELU -> 1x1 + residual), upsample
  ln_kernel<<<MTOT / 256, 256, 0, stream>>>(prm_f, ln_g, ln_b, pnT_f, pnS_f);
  dw3x3_kernel<<<ELEM_BLKS, 256, 0, stream>>>(pnS_f, dsc_dw, dsc_dwb, dsc_h);
  gemm64_kernel<true, true, false><<<GEMM_GRID, 256, 0, stream>>>(
      dsc_h, dsc_pw, dsc_pwb, pnT_f, dout_f, nullptr);
  up2_kernel<<<(BB * CC * HH * WW) / 256, 256, 0, stream>>>(dout_f, out);
}
